// STHGAT_76124000354344
// MI455X (gfx1250) — matrix operand order repaired and checked by reading, not compiled, not run on hardware
//
#include <hip/hip_runtime.h>
#include <hip/hip_bf16.h>

// ---------------- model constants ----------------
static constexpr int kB = 4, kT = 24, kN = 512, kFin = 17;
static constexpr int kHid = 128, kHeads = 4, kGatL = 2, kTempL = 2;
static constexpr int kM = 64, kE = 16384, kDH = 32;
static constexpr int kZ = kB * kN;              // 2048 temporal "rows"
static constexpr int kHB = kHeads * kB;         // 16 GAT batches per t

typedef __attribute__((ext_vector_type(16))) _Float16 v16h;
typedef __attribute__((ext_vector_type(8)))  _Float16 v8h;
typedef __attribute__((ext_vector_type(8)))  float    v8f;

// ---------------- helpers ----------------
__device__ __forceinline__ float act_apply(float x, int act) {
  if (act == 1) return x > 0.f ? x : 0.f;                    // relu
  if (act == 2) return x > 0.f ? x : (__expf(x) - 1.f);      // elu
  return x;
}

__device__ __forceinline__ v8h cvt8(float4 a, float4 b) {
  v8h r;
  r[0] = (_Float16)a.x; r[1] = (_Float16)a.y; r[2] = (_Float16)a.z; r[3] = (_Float16)a.w;
  r[4] = (_Float16)b.x; r[5] = (_Float16)b.y; r[6] = (_Float16)b.z; r[7] = (_Float16)b.w;
  return r;
}

__device__ __forceinline__ v16h cat16(v8h lo, v8h hi) {
  return __builtin_shufflevector(lo, hi, 0, 1, 2, 3, 4, 5, 6, 7,
                                         8, 9, 10, 11, 12, 13, 14, 15);
}

// ---------------- generic batched WMMA GEMM ----------------
// C[z] (+)= act( A[batchA] * B[batchB] + bias ), f32 in/out, f16 WMMA compute.
// batchA = aMod ? z % aMod : z ;  batchB = z / bDiv.
// Requires M%64==0, N%64==0. GUARD handles K not multiple of 32 (scalar path).
// Block: 128 thr = 4 waves; tile 64x64; each wave owns 32x32 (2x2 WMMA accs).
// A staged row-major (sA[m][k]); B staged TRANSPOSED (sBt[n][k]) so both
// fragment types are built from contiguous ds_load_b128 pairs.
template <bool GUARD, bool AHALF>
__global__ __launch_bounds__(128) void wmma_gemm(
    const void* __restrict__ Av, int lda, long long strideA, int aMod,
    const float* __restrict__ Bm, int ldb, long long strideB, int bDiv,
    float* __restrict__ C, int ldc, long long strideC,
    const float* __restrict__ bias,
    int Mdim, int Ndim, int Kdim, int act, int accum)
{
  __shared__ _Float16 sA[64][48];    // rows 96B -> 16B-aligned b128 slots
  __shared__ _Float16 sBt[64][48];   // transposed: [n][k]

  const int z = blockIdx.z;
  const long long aOff = (long long)(aMod ? (z % aMod) : z) * strideA;
  const long long bOff = (long long)(z / bDiv) * strideB;
  const long long cOff = (long long)z * strideC;

  const int m0 = blockIdx.y * 64;
  const int n0 = blockIdx.x * 64;
  const int tid = threadIdx.x;
  const int wave = tid >> 5, lane = tid & 31;
  const int wr = wave >> 1, wc = wave & 1;
  const int lm = lane & 15, lg = lane >> 4;

  const float*    Af = (const float*)Av;
  const _Float16* Ah = (const _Float16*)Av;

  v8f acc00 = {}, acc01 = {}, acc10 = {}, acc11 = {};

  for (int k0 = 0; k0 < Kdim; k0 += 32) {
    if (GUARD) {
      // scalar staging, per-element K bound (only used for the K=17 GEMM)
      for (int idx = tid; idx < 64 * 32; idx += 128) {
        int r = idx >> 5, c = idx & 31;
        int gk = k0 + c;
        float v = 0.f;
        if (gk < Kdim) v = Af[aOff + (long long)(m0 + r) * lda + gk];
        sA[r][c] = (_Float16)v;
      }
      for (int idx = tid; idx < 32 * 64; idx += 128) {
        int r = idx >> 6, c = idx & 63;
        int gk = k0 + r;
        float v = (gk < Kdim) ? Bm[bOff + (long long)gk * ldb + n0 + c] : 0.f;
        sBt[c][r] = (_Float16)v;
      }
    } else {
      // vector staging, fully in-bounds (M%64==0, N%64==0, K%32==0)
      {
        int r = tid >> 1, cb = (tid & 1) * 16;
        if (AHALF) {
          const v8h* src = (const v8h*)(Ah + aOff + (long long)(m0 + r) * lda + k0 + cb);
          *(v8h*)&sA[r][cb]     = src[0];
          *(v8h*)&sA[r][cb + 8] = src[1];
        } else {
          const float4* src = (const float4*)(Af + aOff + (long long)(m0 + r) * lda + k0 + cb);
          float4 f0 = src[0], f1 = src[1], f2 = src[2], f3 = src[3];
          *(v8h*)&sA[r][cb]     = cvt8(f0, f1);
          *(v8h*)&sA[r][cb + 8] = cvt8(f2, f3);
        }
      }
      {
        int kk = tid >> 2, cb = (tid & 3) * 16;
        const float4* src = (const float4*)(Bm + bOff + (long long)(k0 + kk) * ldb + n0 + cb);
        float4 f0 = src[0], f1 = src[1], f2 = src[2], f3 = src[3];
        v8h h0v = cvt8(f0, f1), h1v = cvt8(f2, f3);
#pragma unroll
        for (int j = 0; j < 8; ++j) {
          sBt[cb + j][kk]     = h0v[j];
          sBt[cb + 8 + j][kk] = h1v[j];
        }
      }
    }
    __syncthreads();

    // fragments per ISA 7.12.2 wave32 layouts, all as contiguous b128 LDS reads
    const int ar0 = wr * 32 + lm, ar1 = ar0 + 16;
    v16h a0 = cat16(*(const v8h*)&sA[ar0][lg * 8], *(const v8h*)&sA[ar0][16 + lg * 8]);
    v16h a1 = cat16(*(const v8h*)&sA[ar1][lg * 8], *(const v8h*)&sA[ar1][16 + lg * 8]);
    const int bc0 = wc * 32 + lm, bc1 = bc0 + 16;
    v16h b0 = cat16(*(const v8h*)&sBt[bc0][lg * 8], *(const v8h*)&sBt[bc0][16 + lg * 8]);
    v16h b1 = cat16(*(const v8h*)&sBt[bc1][lg * 8], *(const v8h*)&sBt[bc1][16 + lg * 8]);

    acc00 = __builtin_amdgcn_wmma_f32_16x16x32_f16(false, a0, false, b0, (short)0, acc00, false, false);
    acc01 = __builtin_amdgcn_wmma_f32_16x16x32_f16(false, a0, false, b1, (short)0, acc01, false, false);
    acc10 = __builtin_amdgcn_wmma_f32_16x16x32_f16(false, a1, false, b0, (short)0, acc10, false, false);
    acc11 = __builtin_amdgcn_wmma_f32_16x16x32_f16(false, a1, false, b1, (short)0, acc11, false, false);
    __syncthreads();
  }

  // epilogue: C/D layout m = v + 8*(lane/16), n = lane%16 ; M,N aligned
  const int gn0 = n0 + wc * 32 + lm;
  const int gn1 = gn0 + 16;
  float bs0 = bias ? bias[gn0] : 0.f;
  float bs1 = bias ? bias[gn1] : 0.f;
#pragma unroll
  for (int v = 0; v < 8; ++v) {
    int gm0 = m0 + wr * 32 + lg * 8 + v;
    int gm1 = gm0 + 16;
    long long r0 = cOff + (long long)gm0 * ldc;
    long long r1 = cOff + (long long)gm1 * ldc;
    float x;
    x = acc00[v] + bs0; if (accum) x += C[r0 + gn0]; C[r0 + gn0] = act_apply(x, act);
    x = acc01[v] + bs1; if (accum) x += C[r0 + gn1]; C[r0 + gn1] = act_apply(x, act);
    x = acc10[v] + bs0; if (accum) x += C[r1 + gn0]; C[r1 + gn0] = act_apply(x, act);
    x = acc11[v] + bs1; if (accum) x += C[r1 + gn1]; C[r1 + gn1] = act_apply(x, act);
  }
}

// ---------------- hypergraph laplacian ----------------
__global__ void k_hg_prep(const float* __restrict__ H, float* dv, float* de) {
  int i = blockIdx.x * blockDim.x + threadIdx.x;
  if (i < kN) { float s = 0.f; for (int m = 0; m < kM; ++m) s += H[i * kM + m]; dv[i] = rsqrtf(s); }
  if (i < kM) { float s = 0.f; for (int n = 0; n < kN; ++n) s += H[n * kM + i]; de[i] = 1.f / s; }
}

__global__ void k_lap(const float* __restrict__ H, const float* __restrict__ dv,
                      const float* __restrict__ de, float* __restrict__ lap) {
  int idx = blockIdx.x * blockDim.x + threadIdx.x;
  if (idx >= kN * kN) return;
  int i = idx / kN, j = idx % kN;
  float s = 0.f;
  for (int m = 0; m < kM; ++m) s += H[i * kM + m] * de[m] * H[j * kM + m];
  lap[idx] = dv[i] * s * dv[j];
}

// ---------------- small data movers ----------------
__global__ void k_slice_t(const float* __restrict__ h0, float* __restrict__ h, int t) {
  int idx = blockIdx.x * blockDim.x + threadIdx.x;
  if (idx >= kB * kN * kHid) return;
  int f = idx % kHid, n = (idx / kHid) % kN, b = idx / (kN * kHid);
  h[idx] = h0[(((long long)(b * kT + t)) * kN + n) * kHid + f];
}

__global__ void k_store_t(const float* __restrict__ src, float* __restrict__ y, int t) {
  int idx = blockIdx.x * blockDim.x + threadIdx.x;
  if (idx >= kB * kN * kHid) return;
  int f = idx % kHid, n = (idx / kHid) % kN, b = idx / (kN * kHid);
  y[(((long long)(b * kN + n)) * kT + t) * kHid + f] = src[idx];
}

__global__ void k_copy_f32(float* __restrict__ dst, const float* __restrict__ src, int n) {
  int i = blockIdx.x * blockDim.x + threadIdx.x;
  if (i < n) dst[i] = src[i];
}

__global__ void k_fill_u32(unsigned int* __restrict__ p, int n) {
  int i = blockIdx.x * blockDim.x + threadIdx.x;
  if (i < n) p[i] = 0u;
}

// ---------------- GAT attention pieces ----------------
__global__ void k_alpha(const float* __restrict__ Wh, const float* __restrict__ ga,
                        float* __restrict__ asrc, float* __restrict__ adst) {
  int r = blockIdx.x * blockDim.x + threadIdx.x;
  if (r >= kHB * kN) return;
  int z = r / kN, hd = z / kB;
  const float* w  = Wh + (long long)r * kHid;
  const float* a1 = ga + hd * 2 * kHid;
  const float* a2 = a1 + kHid;
  float s1 = 0.f, s2 = 0.f;
  for (int f = 0; f < kHid; ++f) { float v = w[f]; s1 += v * a1[f]; s2 += v * a2[f]; }
  asrc[r] = s1; adst[r] = s2;
}

__global__ void k_scatter(const int* __restrict__ eidx, const float* __restrict__ ew,
                          const float* __restrict__ asrc, const float* __restrict__ adst,
                          _Float16* __restrict__ att) {
  int idx = blockIdx.x * blockDim.x + threadIdx.x;
  if (idx >= kE * kHB) return;
  int e = idx % kE, z = idx / kE;
  int s = eidx[e], d = eidx[kE + e];
  float v = asrc[z * kN + s] + adst[z * kN + d];
  v = (v > 0.f) ? v : 0.2f * v;                 // leaky_relu(0.2)
  v *= ew[e];
  att[(long long)z * kN * kN + (long long)s * kN + d] = (_Float16)v;
}

// masked row softmax (mask = stored value == 0); one wave per row of 512
__global__ __launch_bounds__(128) void k_softmax_att(_Float16* __restrict__ att) {
  int row  = blockIdx.x * 4 + (threadIdx.x >> 5);
  int lane = threadIdx.x & 31;
  if (row >= kHB * kN) return;
  _Float16* p = att + (long long)row * kN;
  float v[16]; bool msk[16];
  float mx = -INFINITY;
#pragma unroll
  for (int i = 0; i < 16; ++i) {
    float x = (float)p[lane + 32 * i];
    v[i] = x; msk[i] = (x == 0.f);
    if (!msk[i]) mx = fmaxf(mx, x);
  }
  for (int o = 16; o; o >>= 1) mx = fmaxf(mx, __shfl_xor(mx, o, 32));
  float sum = 0.f;
#pragma unroll
  for (int i = 0; i < 16; ++i) {
    float e = msk[i] ? 0.f : __expf(v[i] - mx);
    v[i] = e; sum += e;
  }
  for (int o = 16; o; o >>= 1) sum += __shfl_xor(sum, o, 32);
  float inv = (sum > 0.f) ? 1.f / sum : 0.f;
#pragma unroll
  for (int i = 0; i < 16; ++i) p[lane + 32 * i] = (_Float16)(v[i] * inv);
}

// ---------------- temporal causal attention (T=24, DH=32) ----------------
__global__ __launch_bounds__(128) void k_temporal_attn(const float* __restrict__ qkv,
                                                       float* __restrict__ obuf) {
  int z = blockIdx.x, hd = blockIdx.y, tid = threadIdx.x;
  __shared__ float sq[24][32], sk[24][32], sv[24][32], sp[24][24];
  for (int idx = tid; idx < 24 * 32; idx += 128) {
    int t = idx >> 5, d = idx & 31;
    const float* base = qkv + ((long long)(z * kT + t)) * (3 * kHid) + hd * kDH + d;
    sq[t][d] = base[0]; sk[t][d] = base[kHid]; sv[t][d] = base[2 * kHid];
  }
  __syncthreads();
  const float scale = 0.1767766953f;            // 1/sqrt(32)
  for (int idx = tid; idx < 24 * 24; idx += 128) {
    int t = idx / 24, s = idx % 24;
    float acc = -INFINITY;
    if (s <= t) {
      acc = 0.f;
      for (int d = 0; d < 32; ++d) acc += sq[t][d] * sk[s][d];
      acc *= scale;
    }
    sp[t][s] = acc;
  }
  __syncthreads();
  if (tid < 24) {
    int t = tid; float mx = -INFINITY;
    for (int s = 0; s <= t; ++s) mx = fmaxf(mx, sp[t][s]);
    float sum = 0.f;
    for (int s = 0; s < 24; ++s) {
      float e = (s <= t) ? __expf(sp[t][s] - mx) : 0.f;
      sp[t][s] = e; sum += e;
    }
    float inv = 1.f / sum;
    for (int s = 0; s < 24; ++s) sp[t][s] *= inv;
  }
  __syncthreads();
  for (int idx = tid; idx < 24 * 32; idx += 128) {
    int t = idx >> 5, d = idx & 31;
    float acc = 0.f;
    for (int s = 0; s < 24; ++s) acc += sp[t][s] * sv[s][d];
    obuf[((long long)(z * kT + t)) * kHid + hd * kDH + d] = acc;
  }
}

// layernorm over HID=128; one wave per row
__global__ __launch_bounds__(128) void k_layernorm(float* __restrict__ y,
                                                   const float* __restrict__ g,
                                                   const float* __restrict__ bb) {
  int row = blockIdx.x * 4 + (threadIdx.x >> 5);
  int lane = threadIdx.x & 31;
  if (row >= kZ * kT) return;
  float* p = y + (long long)row * kHid;
  float x0 = p[lane], x1 = p[lane + 32], x2 = p[lane + 64], x3 = p[lane + 96];
  float s = x0 + x1 + x2 + x3;
  for (int o = 16; o; o >>= 1) s += __shfl_xor(s, o, 32);
  float mu = s * (1.0f / kHid);
  float d0 = x0 - mu, d1 = x1 - mu, d2 = x2 - mu, d3 = x3 - mu;
  float vs = d0 * d0 + d1 * d1 + d2 * d2 + d3 * d3;
  for (int o = 16; o; o >>= 1) vs += __shfl_xor(vs, o, 32);
  float inv = rsqrtf(vs * (1.0f / kHid) + 1e-5f);
  p[lane]      = d0 * inv * g[lane]      + bb[lane];
  p[lane + 32] = d1 * inv * g[lane + 32] + bb[lane + 32];
  p[lane + 64] = d2 * inv * g[lane + 64] + bb[lane + 64];
  p[lane + 96] = d3 * inv * g[lane + 96] + bb[lane + 96];
}

__global__ void k_mean_T(const float* __restrict__ y, float* __restrict__ zb) {
  int idx = blockIdx.x * blockDim.x + threadIdx.x;
  if (idx >= kZ * kHid) return;
  int z = idx / kHid, f = idx % kHid;
  float s = 0.f;
  for (int t = 0; t < kT; ++t) s += y[((long long)z * kT + t) * kHid + f];
  zb[idx] = s * (1.0f / kT);
}

__global__ void k_final(const float* __restrict__ z1, const float* __restrict__ o2w,
                        const float* __restrict__ o2b, float* __restrict__ out) {
  int i = blockIdx.x * blockDim.x + threadIdx.x;
  if (i >= kZ) return;
  float s = 0.f;
  for (int f = 0; f < 64; ++f) s += z1[i * 64 + f] * o2w[f];
  out[i] = s + o2b[0];
}

// ---------------- host-side launcher ----------------
static inline void gemm(hipStream_t st,
                        const void* A, int aHalf, int guard, int lda, long long sA, int aMod,
                        const float* Bm, int ldb, long long sB, int bDiv,
                        float* C, int ldc, long long sC, const float* bias,
                        int M, int Nd, int K, int act, int accum, int batches) {
  dim3 g((Nd + 63) / 64, (M + 63) / 64, batches);
  if (guard)
    wmma_gemm<true, false><<<g, dim3(128), 0, st>>>(A, lda, sA, aMod, Bm, ldb, sB, bDiv,
                                                    C, ldc, sC, bias, M, Nd, K, act, accum);
  else if (aHalf)
    wmma_gemm<false, true><<<g, dim3(128), 0, st>>>(A, lda, sA, aMod, Bm, ldb, sB, bDiv,
                                                    C, ldc, sC, bias, M, Nd, K, act, accum);
  else
    wmma_gemm<false, false><<<g, dim3(128), 0, st>>>(A, lda, sA, aMod, Bm, ldb, sB, bDiv,
                                                     C, ldc, sC, bias, M, Nd, K, act, accum);
}

extern "C" void kernel_launch(void* const* d_in, const int* in_sizes, int n_in,
                              void* d_out, int out_size, void* d_ws, size_t ws_size,
                              hipStream_t stream) {
  const float* x      = (const float*)d_in[0];
  const int*   eidx   = (const int*)d_in[1];
  const float* ew     = (const float*)d_in[2];
  const float* H      = (const float*)d_in[3];
  const float* w_in   = (const float*)d_in[4];
  const float* b_in   = (const float*)d_in[5];
  const float* gat_W  = (const float*)d_in[6];
  const float* gat_a  = (const float*)d_in[7];
  const float* gat_pw = (const float*)d_in[8];
  const float* gat_pb = (const float*)d_in[9];
  const float* hg_w   = (const float*)d_in[10];
  const float* hg_b   = (const float*)d_in[11];
  const float* fu_w   = (const float*)d_in[12];
  const float* fu_b   = (const float*)d_in[13];
  const float* t_in_w = (const float*)d_in[14];
  const float* t_in_b = (const float*)d_in[15];
  const float* t_out_w= (const float*)d_in[16];
  const float* t_out_b= (const float*)d_in[17];
  const float* t_ln_g = (const float*)d_in[18];
  const float* t_ln_b = (const float*)d_in[19];
  const float* o1_w   = (const float*)d_in[20];
  const float* o1_b   = (const float*)d_in[21];
  const float* o2_w   = (const float*)d_in[22];
  const float* o2_b   = (const float*)d_in[23];
  float* out = (float*)d_out;

  // ---- workspace layout (persistent + overlaid union region) ----
  char* ws = (char*)d_ws;
  size_t off = 0;
  auto alloc = [&](size_t bytes) -> char* {
    char* p = ws + off; off += (bytes + 255) & ~(size_t)255; return p;
  };
  float* lap  = (float*)alloc(sizeof(float) * (size_t)kN * kN);
  float* dv   = (float*)alloc(sizeof(float) * kN);
  float* de   = (float*)alloc(sizeof(float) * kM);
  float* ybuf = (float*)alloc(sizeof(float) * (size_t)kZ * kT * kHid);
  float* zb   = (float*)alloc(sizeof(float) * (size_t)kZ * kHid);
  float* z1   = (float*)alloc(sizeof(float) * (size_t)kZ * 64);
  char* un = ws + off;
  // GAT-stage overlay
  float* h0    = (float*)un;
  float* hA    = h0    + (size_t)kB * kT * kN * kHid;
  float* hBp   = hA    + (size_t)kB * kN * kHid;
  float* tmpb  = hBp   + (size_t)kB * kN * kHid;
  float* hhb   = tmpb  + (size_t)kB * kN * kHid;
  float* fuseb = hhb   + (size_t)kB * kN * kHid;
  float* Wh    = fuseb + (size_t)kB * kN * kHid;
  float* asrc  = Wh    + (size_t)kHB * kN * kHid;
  float* adst  = asrc  + (size_t)kHB * kN;
  float* hp    = adst  + (size_t)kHB * kN;
  _Float16* att = (_Float16*)(hp + (size_t)kHB * kN * kHid);
  // temporal-stage overlay (GAT scratch is dead by then)
  float* qkv  = (float*)un;
  float* obuf = qkv + (size_t)kZ * kT * 3 * kHid;

  const int BNH = kB * kN * kHid;             // 262144
  const int attWords = (kHB * kN * kN) / 2;   // f16 buffer as u32 words

  // ---- hypergraph laplacian ----
  k_hg_prep<<<2, 256, 0, stream>>>(H, dv, de);
  k_lap<<<(kN * kN + 255) / 256, 256, 0, stream>>>(H, dv, de, lap);

  // ---- input projection: h0 = x @ w_in + b_in  (K=17 -> guarded path) ----
  gemm(stream, x, 0, 1, kFin, 0, 1, w_in, kHid, 0, 1,
       h0, kHid, 0, b_in, kB * kT * kN, kHid, kFin, 0, 0, 1);

  // ---- spatial stage: T independent steps (scan carry is unused) ----
  for (int t = 0; t < kT; ++t) {
    k_slice_t<<<(BNH + 255) / 256, 256, 0, stream>>>(h0, hA, t);
    float* hcur = hA; float* hnext = hBp;
    for (int l = 0; l < kGatL; ++l) {
      // Wh[z=hd*B+b] = h[b] @ W[l,hd] : 16 batches of 512x128x128
      gemm(stream, hcur, 0, 0, kHid, (long long)kN * kHid, kB,
           gat_W + (size_t)l * kHeads * kHid * kHid, kHid, (long long)kHid * kHid, kB,
           Wh, kHid, (long long)kN * kHid, nullptr, kN, kHid, kHid, 0, 0, kHB);
      k_alpha<<<(kHB * kN + 255) / 256, 256, 0, stream>>>(
          Wh, gat_a + (size_t)l * kHeads * 2 * kHid, asrc, adst);
      k_fill_u32<<<(attWords + 255) / 256, 256, 0, stream>>>((unsigned int*)att, attWords);
      k_scatter<<<(kE * kHB + 255) / 256, 256, 0, stream>>>(eidx, ew, asrc, adst, att);
      k_softmax_att<<<(kHB * kN) / 4, 128, 0, stream>>>(att);
      // hp[z] = elu(att[z] @ Wh[z]) : 16 batches of 512x512x128, A is f16
      gemm(stream, att, 1, 0, kN, (long long)kN * kN, 0,
           Wh, kHid, (long long)kN * kHid, 1,
           hp, kHid, (long long)kN * kHid, nullptr, kN, kHid, kN, 2, 0, kHB);
      // residual + per-head projection accumulation
      k_copy_f32<<<(BNH + 255) / 256, 256, 0, stream>>>(hnext, hcur, BNH);
      for (int hd = 0; hd < kHeads; ++hd) {
        gemm(stream, hp + (size_t)hd * kB * kN * kHid, 0, 0, kHid, (long long)kN * kHid, 0,
             gat_pw + (size_t)l * kHeads * kHid * kHid + (size_t)hd * kHid * kHid,
             kHid, 0, 1 << 20,
             hnext, kHid, (long long)kN * kHid,
             (hd == 0) ? gat_pb + (size_t)l * kHid : nullptr,
             kN, kHid, kHid, 0, 1, kB);
      }
      float* sw = hcur; hcur = hnext; hnext = sw;
    }
    // hh = lap @ (h @ hg_w + hg_b)
    gemm(stream, hcur, 0, 0, kHid, 0, 1, hg_w, kHid, 0, 1,
         tmpb, kHid, 0, hg_b, kB * kN, kHid, kHid, 0, 0, 1);
    gemm(stream, lap, 0, 0, kN, 0, 1,
         tmpb, kHid, (long long)kN * kHid, 1,
         hhb, kHid, (long long)kN * kHid, nullptr, kN, kHid, kN, 0, 0, kB);
    // h = relu([h, hh] @ fu_w + fu_b) as two K=128 GEMMs
    gemm(stream, hcur, 0, 0, kHid, 0, 1, fu_w, kHid, 0, 1,
         fuseb, kHid, 0, fu_b, kB * kN, kHid, kHid, 0, 0, 1);
    gemm(stream, hhb, 0, 0, kHid, 0, 1, fu_w + (size_t)kHid * kHid, kHid, 0, 1,
         fuseb, kHid, 0, nullptr, kB * kN, kHid, kHid, 1, 1, 1);
    k_store_t<<<(BNH + 255) / 256, 256, 0, stream>>>(fuseb, ybuf, t);
  }

  // ---- temporal stage ----
  for (int l = 0; l < kTempL; ++l) {
    gemm(stream, ybuf, 0, 0, kHid, 0, 1,
         t_in_w + (size_t)l * kHid * 3 * kHid, 3 * kHid, 0, 1,
         qkv, 3 * kHid, 0, t_in_b + (size_t)l * 3 * kHid,
         kZ * kT, 3 * kHid, kHid, 0, 0, 1);
    k_temporal_attn<<<dim3(kZ, kHeads), 128, 0, stream>>>(qkv, obuf);
    gemm(stream, obuf, 0, 0, kHid, 0, 1,
         t_out_w + (size_t)l * kHid * kHid, kHid, 0, 1,
         ybuf, kHid, 0, t_out_b + (size_t)l * kHid,
         kZ * kT, kHid, kHid, 0, 1, 1);
    k_layernorm<<<(kZ * kT) / 4, 128, 0, stream>>>(ybuf, t_ln_g + l * kHid, t_ln_b + l * kHid);
  }

  // ---- output head ----
  k_mean_T<<<(kZ * kHid + 255) / 256, 256, 0, stream>>>(ybuf, zb);
  gemm(stream, zb, 0, 0, kHid, 0, 1, o1_w, 64, 0, 1,
       z1, 64, 0, o1_b, kZ, 64, kHid, 1, 0, 1);
  k_final<<<(kZ + 255) / 256, 256, 0, stream>>>(z1, o2_w, o2_b, out);
}
